// Transformer_news_classifier_2_29918742184491
// MI455X (gfx1250) — matrix operand order repaired and checked by reading, not compiled, not run on hardware
//
#include <hip/hip_runtime.h>
#include <hip/hip_bf16.h>

// ---------------------------------------------------------------------------
// Transformer news classifier on MI455X (gfx1250, wave32, WMMA).
// All GEMMs: f16 inputs, f32 accumulate via v_wmma_f32_16x16x32_f16.
// ---------------------------------------------------------------------------

typedef __attribute__((ext_vector_type(16))) _Float16 v16h;
typedef __attribute__((ext_vector_type(8)))  _Float16 v8h;
typedef __attribute__((ext_vector_type(8)))  float    v8f;

#define BB   32
#define SS   512
#define DDIM 768
#define HH   768
#define NHH  4
#define HDD  192
#define MM   (BB * SS)          // 16384 rows
#define CC   10

// ---------------------------------------------------------------------------
// Helpers: WMMA fragment loads, contiguous k order used for both A and B.
// A-frag (16x32 f16, MxK): lane L holds row L%16, K = (L/16)*16 + {0..15}
//                          in halves 0..15 (contiguous, matching B-frag).
// B-frag (32x16 f16, KxN): lane L holds col L%16, K = (L/16)*16 + {0..15},
//                          contiguous 32 bytes when the "N" rows are K-major.
// ---------------------------------------------------------------------------

__device__ __forceinline__ v16h load_afrag(const _Float16* p /* at (m0,k0), row r & hs*16 pre-applied */) {
  v8h lo = *(const v8h*)(p);
  v8h hi = *(const v8h*)(p + 8);
  return __builtin_shufflevector(lo, hi, 0,1,2,3,4,5,6,7,8,9,10,11,12,13,14,15);
}

__device__ __forceinline__ v8f wmma_f16(v16h a, v16h b, v8f c) {
  return __builtin_amdgcn_wmma_f32_16x16x32_f16(false, a, false, b, (short)0, c, false, false);
}

// ---------------------------------------------------------------------------
// Kernel 1: x + sinusoidal PE, emit f16.
// ---------------------------------------------------------------------------
__global__ __launch_bounds__(256) void pe_add_kernel(const float* __restrict__ x,
                                                     _Float16* __restrict__ out) {
  int idx = blockIdx.x * 256 + threadIdx.x;
  if (idx >= MM * DDIM) return;
  int d = idx % DDIM;
  int s = (idx / DDIM) % SS;
  int ieven = d & ~1;
  float angle = (float)s * __powf(10000.0f, -2.0f * (float)ieven / (float)DDIM);
  float pe = (d & 1) ? __cosf(angle) : __sinf(angle);
  out[idx] = (_Float16)(x[idx] + pe);
}

// ---------------------------------------------------------------------------
// Kernel 2: generic GEMM  Y[M,N] = X[M,K] @ W[N,K]^T + bias, optional relu.
// One wave computes a 32x64 tile of Y: 2 A-frags x 4 B-frags -> 8 accumulators.
// ~21 FLOP/byte from L2 per 32-K step; ~130 live VGPRs.
// ---------------------------------------------------------------------------
__global__ __launch_bounds__(256) void gemm_wmma(const _Float16* __restrict__ X,
                                                 const _Float16* __restrict__ W,
                                                 const float* __restrict__ bias,
                                                 float* __restrict__ Y32,
                                                 _Float16* __restrict__ Y16,
                                                 int M, int N, int K, int relu) {
  int wave = blockIdx.x * (blockDim.x >> 5) + (threadIdx.x >> 5);
  int lane = threadIdx.x & 31;
  int ntiles = N / 64;
  int total = (M / 32) * ntiles;
  if (wave >= total) return;
  int mt = wave / ntiles;
  int nt = wave % ntiles;

  int r  = lane & 15;
  int hs = lane >> 4;

  const _Float16* ax0 = X + ((size_t)(mt * 32 + r)) * K + hs * 16;
  const _Float16* ax1 = ax0 + (size_t)16 * K;
  const _Float16* wp  = W + ((size_t)(nt * 64 + r)) * K + hs * 16;
  size_t wstep = (size_t)16 * K;

  v8f acc[2][4];
  v8f z = {};
#pragma unroll
  for (int i = 0; i < 2; ++i)
#pragma unroll
    for (int t = 0; t < 4; ++t) acc[i][t] = z;

  for (int k0 = 0; k0 < K; k0 += 32) {
    __builtin_prefetch(ax0 + k0 + 64, 0, 3);   // global_prefetch_b8, WGP scope
    __builtin_prefetch(ax1 + k0 + 64, 0, 3);
    v16h a0 = load_afrag(ax0 + k0);
    v16h a1 = load_afrag(ax1 + k0);
    v16h b0 = *(const v16h*)(wp + k0);
    v16h b1 = *(const v16h*)(wp + wstep + k0);
    v16h b2 = *(const v16h*)(wp + 2 * wstep + k0);
    v16h b3 = *(const v16h*)(wp + 3 * wstep + k0);
    acc[0][0] = wmma_f16(a0, b0, acc[0][0]);
    acc[1][0] = wmma_f16(a1, b0, acc[1][0]);
    acc[0][1] = wmma_f16(a0, b1, acc[0][1]);
    acc[1][1] = wmma_f16(a1, b1, acc[1][1]);
    acc[0][2] = wmma_f16(a0, b2, acc[0][2]);
    acc[1][2] = wmma_f16(a1, b2, acc[1][2]);
    acc[0][3] = wmma_f16(a0, b3, acc[0][3]);
    acc[1][3] = wmma_f16(a1, b3, acc[1][3]);
  }

  int rbase = hs * 8;
#pragma unroll
  for (int i = 0; i < 2; ++i) {
#pragma unroll
    for (int t = 0; t < 4; ++t) {
      int n = nt * 64 + t * 16 + r;
      float bv = bias ? bias[n] : 0.0f;
#pragma unroll
      for (int v = 0; v < 8; ++v) {
        float y = acc[i][t][v] + bv;
        if (relu) y = fmaxf(y, 0.0f);
        size_t off = (size_t)(mt * 32 + i * 16 + rbase + v) * N + n;
        if (Y32) Y32[off] = y;
        if (Y16) Y16[off] = (_Float16)y;
      }
    }
  }
}

// ---------------------------------------------------------------------------
// Kernel 3: transpose V heads: v2[B*S, H] -> vt[B][NH][HD][S]
// ---------------------------------------------------------------------------
__global__ __launch_bounds__(256) void transpose_v_kernel(const _Float16* __restrict__ v2,
                                                          _Float16* __restrict__ vt) {
  int idx = blockIdx.x * 256 + threadIdx.x;
  if (idx >= BB * NHH * HDD * SS) return;
  int s = idx % SS;
  int d = (idx / SS) % HDD;
  int h = (idx / (SS * HDD)) % NHH;
  int b = idx / (SS * HDD * NHH);
  vt[idx] = v2[((size_t)(b * SS + s)) * HH + h * HDD + d];
}

// ---------------------------------------------------------------------------
// Kernel 4: flash attention. Each wave: 16 query rows of one (b,h).
// Q,K: [B*S, H] f16 (head slice). Vt: [B][NH][HD][S] f16. ctx: [B*S, H] f16.
// ---------------------------------------------------------------------------
__global__ __launch_bounds__(128) void attn_wmma(const _Float16* __restrict__ Q,
                                                 const _Float16* __restrict__ Km,
                                                 const _Float16* __restrict__ Vt,
                                                 _Float16* __restrict__ ctx) {
  const int w    = threadIdx.x >> 5;
  const int lane = threadIdx.x & 31;
  const int gw   = blockIdx.x * 4 + w;        // 0..4095 (grid sized exactly)
  const int qt = gw & 31;
  const int h  = (gw >> 5) & 3;
  const int b  = gw >> 7;

  __shared__ __align__(32) float    sS[4][16][32];
  __shared__ __align__(32) _Float16 sP[4][16][32];
  __shared__ float sA[4][16];
  __shared__ float sL[4][16];

  const int r  = lane & 15;
  const int hs = lane >> 4;
  const int rb = hs * 8;

  const _Float16* qb = Q  + ((size_t)(b * SS + qt * 16 + r)) * HH + h * HDD + hs * 16;
  const _Float16* kb = Km + ((size_t)(b * SS + r)) * HH + h * HDD + hs * 16;
  const _Float16* vb = Vt + ((size_t)(b * NHH + h)) * HDD * SS + (size_t)r * SS + hs * 16;

  v8f acc[12];
  v8f z = {};
#pragma unroll
  for (int t = 0; t < 12; ++t) acc[t] = z;
  float m_r = -3.0e38f, l_r = 0.0f;
  const float scale = 0.0721687836487032f;    // 1/sqrt(192)

  for (int kc = 0; kc < SS; kc += 32) {
    // ---- scores tile 16x32 = Q(16xHD) x K^T(HDx32) ----
    v8f c0 = z, c1 = z;
#pragma unroll
    for (int d0 = 0; d0 < HDD; d0 += 32) {
      v16h a  = load_afrag(qb + d0);
      v16h b0 = *(const v16h*)(kb + (size_t)kc * HH + d0);
      v16h b1 = *(const v16h*)(kb + (size_t)(kc + 16) * HH + d0);
      c0 = wmma_f16(a, b0, c0);
      c1 = wmma_f16(a, b1, c1);
    }
#pragma unroll
    for (int v = 0; v < 8; ++v) {
      sS[w][rb + v][r]      = c0[v] * scale;
      sS[w][rb + v][r + 16] = c1[v] * scale;
    }
    __syncthreads();

    // ---- online softmax (lanes 0..15: one query row each) ----
    if (lane < 16) {
      float mc = -3.0e38f;
#pragma unroll
      for (int j = 0; j < 32; ++j) mc = fmaxf(mc, sS[w][lane][j]);
      float mn = fmaxf(m_r, mc);
      float al = __expf(m_r - mn);
      float ps = 0.0f;
#pragma unroll
      for (int j = 0; j < 32; ++j) {
        float p = __expf(sS[w][lane][j] - mn);
        ps += p;
        sP[w][lane][j] = (_Float16)p;
      }
      l_r = l_r * al + ps;
      m_r = mn;
      sA[w][lane] = al;
    }
    __syncthreads();

    // ---- rescale accumulators, then ctx += P(16x32) x V(32xHD) ----
    float av[8];
#pragma unroll
    for (int v = 0; v < 8; ++v) av[v] = sA[w][rb + v];
    v16h pa = load_afrag(&sP[w][r][hs * 16]);
#pragma unroll
    for (int t = 0; t < 12; ++t) {
#pragma unroll
      for (int v = 0; v < 8; ++v) acc[t][v] *= av[v];
      v16h bvf = *(const v16h*)(vb + (size_t)(t * 16) * SS + kc);
      acc[t] = wmma_f16(pa, bvf, acc[t]);
    }
    __syncthreads();
  }

  if (lane < 16) sL[w][lane] = l_r;
  __syncthreads();
  float rl[8];
#pragma unroll
  for (int v = 0; v < 8; ++v) rl[v] = 1.0f / sL[w][rb + v];

  _Float16* cb = ctx + ((size_t)(b * SS + qt * 16)) * HH + h * HDD;
#pragma unroll
  for (int t = 0; t < 12; ++t)
#pragma unroll
    for (int v = 0; v < 8; ++v)
      cb[(size_t)(rb + v) * HH + t * 16 + r] = (_Float16)(acc[t][v] * rl[v]);
}

// ---------------------------------------------------------------------------
// Kernel 5: LayerNorm(A + B) -> out32 (+ optional f16 copy). One block per row.
// ---------------------------------------------------------------------------
__global__ __launch_bounds__(256) void ln_residual(const float* __restrict__ A,
                                                   const float* __restrict__ Bv,
                                                   const float* __restrict__ g,
                                                   const float* __restrict__ bb,
                                                   float* __restrict__ out32,
                                                   _Float16* __restrict__ out16) {
  int row = blockIdx.x;
  int tid = threadIdx.x;
  const float* a = A  + (size_t)row * HH;
  const float* c = Bv + (size_t)row * HH;
  __shared__ float red[256];

  float vals[3];
  float s = 0.0f;
#pragma unroll
  for (int i = 0; i < 3; ++i) {
    int col = tid + i * 256;
    vals[i] = a[col] + c[col];
    s += vals[i];
  }
  red[tid] = s;
  __syncthreads();
  for (int off = 128; off > 0; off >>= 1) {
    if (tid < off) red[tid] += red[tid + off];
    __syncthreads();
  }
  float mean = red[0] * (1.0f / HH);
  __syncthreads();

  float vs = 0.0f;
#pragma unroll
  for (int i = 0; i < 3; ++i) {
    float d = vals[i] - mean;
    vs += d * d;
  }
  red[tid] = vs;
  __syncthreads();
  for (int off = 128; off > 0; off >>= 1) {
    if (tid < off) red[tid] += red[tid + off];
    __syncthreads();
  }
  float rstd = rsqrtf(red[0] * (1.0f / HH) + 1e-5f);

#pragma unroll
  for (int i = 0; i < 3; ++i) {
    int col = tid + i * 256;
    float y = (vals[i] - mean) * rstd * g[col] + bb[col];
    out32[(size_t)row * HH + col] = y;
    if (out16) out16[(size_t)row * HH + col] = (_Float16)y;
  }
}

// ---------------------------------------------------------------------------
// Kernel 6: column-sum over sequence: xsum[b][d] = sum_s x2[b,s,d]
// ---------------------------------------------------------------------------
__global__ __launch_bounds__(256) void colsum_kernel(const float* __restrict__ x2,
                                                     float* __restrict__ xsum) {
  int b = blockIdx.x / 3;
  int d = (blockIdx.x % 3) * 256 + threadIdx.x;
  float s = 0.0f;
  for (int t = 0; t < SS; ++t) s += x2[((size_t)(b * SS + t)) * HH + d];
  xsum[b * HH + d] = s;
}

// ---------------------------------------------------------------------------
// Kernel 7: classifier head (32x10) in fp32.
// ---------------------------------------------------------------------------
__global__ __launch_bounds__(256) void fc2_kernel(const float* __restrict__ xsum,
                                                  const float* __restrict__ w,
                                                  const float* __restrict__ bias,
                                                  float* __restrict__ out) {
  int idx = blockIdx.x * 256 + threadIdx.x;
  if (idx >= BB * CC) return;
  int c = idx % CC, b = idx / CC;
  float s = bias[c];
  for (int d = 0; d < HH; ++d) s += xsum[b * HH + d] * w[c * HH + d];
  out[idx] = s;
}

// ---------------------------------------------------------------------------
// Kernel 8: f32 -> f16 conversion
// ---------------------------------------------------------------------------
__global__ __launch_bounds__(256) void f32_to_f16(const float* __restrict__ in,
                                                  _Float16* __restrict__ out, int n) {
  int idx = blockIdx.x * 256 + threadIdx.x;
  if (idx < n) out[idx] = (_Float16)in[idx];
}

// ---------------------------------------------------------------------------
// Host orchestration
// ---------------------------------------------------------------------------
extern "C" void kernel_launch(void* const* d_in, const int* in_sizes, int n_in,
                              void* d_out, int out_size, void* d_ws, size_t ws_size,
                              hipStream_t stream) {
  const float* x    = (const float*)d_in[0];
  const float* wq   = (const float*)d_in[1];  const float* bq   = (const float*)d_in[2];
  const float* wk   = (const float*)d_in[3];  const float* bk   = (const float*)d_in[4];
  const float* wv   = (const float*)d_in[5];  const float* bv   = (const float*)d_in[6];
  const float* wx   = (const float*)d_in[7];  const float* bx   = (const float*)d_in[8];
  const float* ipw  = (const float*)d_in[9];  const float* ipb  = (const float*)d_in[10];
  const float* opw  = (const float*)d_in[11]; const float* opb  = (const float*)d_in[12];
  const float* f1aw = (const float*)d_in[13]; const float* f1ab = (const float*)d_in[14];
  const float* f1bw = (const float*)d_in[15]; const float* f1bb = (const float*)d_in[16];
  const float* lng  = (const float*)d_in[17]; const float* lnb  = (const float*)d_in[18];
  const float* f2w  = (const float*)d_in[19]; const float* f2b  = (const float*)d_in[20];
  float* out = (float*)d_out;

  char* base = (char*)d_ws;
  size_t cur = 0;
  auto alloc = [&](size_t bytes) -> void* {
    void* p = base + cur;
    cur += (bytes + 255) & ~(size_t)255;
    return p;
  };

  const size_t WSZ  = (size_t)HH * DDIM;          // 589824
  const size_t ASZ  = (size_t)MM * HH;            // 12582912

  _Float16* wq16  = (_Float16*)alloc(WSZ * 2);
  _Float16* wk16  = (_Float16*)alloc(WSZ * 2);
  _Float16* wv16  = (_Float16*)alloc(WSZ * 2);
  _Float16* wx16  = (_Float16*)alloc(WSZ * 2);
  _Float16* ip16  = (_Float16*)alloc(3 * WSZ * 2);
  _Float16* op16  = (_Float16*)alloc(WSZ * 2);
  _Float16* f1a16 = (_Float16*)alloc(WSZ * 2);
  _Float16* f1b16 = (_Float16*)alloc(WSZ * 2);

  _Float16* xpe16 = (_Float16*)alloc(ASZ * 2);
  _Float16* q16   = (_Float16*)alloc(ASZ * 2);
  _Float16* k16   = (_Float16*)alloc(ASZ * 2);
  _Float16* v16   = (_Float16*)alloc(ASZ * 2);
  float*    xr32  = (float*)   alloc(ASZ * 4);
  _Float16* q2    = (_Float16*)alloc(ASZ * 2);
  _Float16* k2    = (_Float16*)alloc(ASZ * 2);
  _Float16* v2    = (_Float16*)alloc(ASZ * 2);
  _Float16* vt16  = (_Float16*)alloc(ASZ * 2);
  float*    attn32= (float*)   alloc(ASZ * 4);
  float*    x1_32 = (float*)   alloc(ASZ * 4);
  float*    xsum  = (float*)   alloc((size_t)BB * HH * 4);

  // buffer reuse (lifetimes verified):
  _Float16* ctx16 = q16;      // q16 dead after in_proj q2
  _Float16* x1_16 = k16;      // k16 dead after in_proj k2
  _Float16* h16   = v16;      // v16 dead after in_proj v2
  float*    ffn32 = attn32;   // attn32 dead after LN1
  float*    x2_32 = xr32;     // xr32 dead after LN1

  auto cvt = [&](const float* src, _Float16* dst, int n) {
    f32_to_f16<<<(n + 255) / 256, 256, 0, stream>>>(src, dst, n);
  };
  auto gemm = [&](const _Float16* X, const _Float16* W, const float* bias,
                  float* Y32, _Float16* Y16, int M, int N, int K, int relu) {
    int waves = (M / 32) * (N / 64);
    gemm_wmma<<<(waves + 7) / 8, 256, 0, stream>>>(X, W, bias, Y32, Y16, M, N, K, relu);
  };

  // 0) weight conversion to f16
  cvt(wq, wq16, (int)WSZ);    cvt(wk, wk16, (int)WSZ);
  cvt(wv, wv16, (int)WSZ);    cvt(wx, wx16, (int)WSZ);
  cvt(ipw, ip16, (int)(3 * WSZ));
  cvt(opw, op16, (int)WSZ);
  cvt(f1aw, f1a16, (int)WSZ); cvt(f1bw, f1b16, (int)WSZ);

  // 1) x + positional encoding -> f16
  pe_add_kernel<<<(MM * DDIM + 255) / 256, 256, 0, stream>>>(x, xpe16);

  // 2) q/k/v/xr projections
  gemm(xpe16, wq16, bq, nullptr, q16, MM, HH, DDIM, 0);
  gemm(xpe16, wk16, bk, nullptr, k16, MM, HH, DDIM, 0);
  gemm(xpe16, wv16, bv, nullptr, v16, MM, HH, DDIM, 0);
  gemm(xpe16, wx16, bx, xr32, nullptr, MM, HH, DDIM, 0);

  // 3) MHA in_proj (split rows of in_proj_w)
  gemm(q16, ip16,               ipb,        nullptr, q2, MM, HH, HH, 0);
  gemm(k16, ip16 + WSZ,         ipb + HH,   nullptr, k2, MM, HH, HH, 0);
  gemm(v16, ip16 + 2 * WSZ,     ipb + 2*HH, nullptr, v2, MM, HH, HH, 0);

  // 4) transpose V per head, then flash attention
  transpose_v_kernel<<<(BB * NHH * HDD * SS + 255) / 256, 256, 0, stream>>>(v2, vt16);
  attn_wmma<<<(BB * NHH * (SS / 16)) / 4, 128, 0, stream>>>(q2, k2, vt16, ctx16);

  // 5) out_proj
  gemm(ctx16, op16, opb, attn32, nullptr, MM, HH, HH, 0);

  // 6) LN1(xr + attn_out)
  ln_residual<<<MM, 256, 0, stream>>>(xr32, attn32, lng, lnb, x1_32, x1_16);

  // 7) FFN: relu(x1 @ fc1a^T) @ fc1b^T
  gemm(x1_16, f1a16, f1ab, nullptr, h16, MM, HH, HH, 1);
  gemm(h16,   f1b16, f1bb, ffn32, nullptr, MM, HH, HH, 0);

  // 8) LN2(x1 + ffn)
  ln_residual<<<MM, 256, 0, stream>>>(x1_32, ffn32, lng, lnb, x2_32, nullptr);

  // 9) sum over sequence, classifier head
  colsum_kernel<<<BB * 3, 256, 0, stream>>>(x2_32, xsum);
  fc2_kernel<<<(BB * CC + 255) / 256, 256, 0, stream>>>(xsum, f2w, f2b, out);
}
